// LSTMRegressor_83476984365063
// MI455X (gfx1250) — compile-verified, matrix operand order checked by reading, not run on hardware
//
#include <hip/hip_runtime.h>

// ---------------------------------------------------------------------------
// Types for CDNA5 WMMA (wave32)
// ---------------------------------------------------------------------------
typedef __bf16 bf16_t;
typedef __attribute__((ext_vector_type(16))) __bf16 v16bf;
typedef __attribute__((ext_vector_type(8)))  __bf16 bf16x8;
typedef __attribute__((ext_vector_type(8)))  float  v8f;

#define B_    128
#define T_    1024
#define IN_   64
#define H_    256
#define OUT_  64
#define NWG   8      // persistent workgroups; each owns 32 hidden units (128 gate cols)
#define THR   512    // 16 waves
#define NGATE 128    // gate columns per WG

__device__ __forceinline__ bf16_t f2bf(float f) {
  union { float f; unsigned u; } v; v.f = f;
  unsigned r = v.u + 0x7FFFu + ((v.u >> 16) & 1u);   // round-to-nearest-even
  union { unsigned short s; bf16_t b; } o; o.s = (unsigned short)(r >> 16);
  return o.b;
}

__device__ __forceinline__ float sigmoidf_(float x) { return 1.0f / (1.0f + __expf(-x)); }
__device__ __forceinline__ float tanhf_(float x) {
  float e = __expf(2.0f * x);
  return (e - 1.0f) / (e + 1.0f);
}

__device__ __forceinline__ v8f vzero8() {
  v8f v = {0.f, 0.f, 0.f, 0.f, 0.f, 0.f, 0.f, 0.f};
  return v;
}

// A/B operand tile loader, bf16 row-major source (LDS or global), leading dim ld.
// 16-bit 16x32 operand lane layout: lane L (L<16): M=L, K in {kb..kb+7, kb+16..kb+23}
// with kb=0; lanes 16..31: same rows, kb=8.  Two 128-bit loads per lane.
__device__ __forceinline__ v16bf ld_bf_tile(const bf16_t* base, int ld, int row0, int k0) {
  int lane = threadIdx.x & 31;
  int m  = lane & 15;
  int kb = (lane >> 4) << 3;           // 0 or 8
  const bf16_t* p = base + (size_t)(row0 + m) * ld + (k0 + kb);
  bf16x8 lo = *(const bf16x8*)(p);
  bf16x8 hi = *(const bf16x8*)(p + 16);
  v16bf r;
#pragma unroll
  for (int i = 0; i < 8; ++i) { r[i] = lo[i]; r[i + 8] = hi[i]; }
  return r;
}

__device__ __forceinline__ v8f wmma_bf16(v16bf a, v16bf b, v8f c) {
  return __builtin_amdgcn_wmma_f32_16x16x32_bf16(false, a, false, b, (short)0, c, false, false);
}

// Monotonic-counter grid barrier across NWG persistent workgroups.
__device__ __forceinline__ void grid_barrier(unsigned* cnt, unsigned& gen) {
  __syncthreads();
  __threadfence();                 // make this thread's global stores visible
  gen += NWG;
  if (threadIdx.x == 0) {
    atomicAdd(cnt, 1u);
    while (atomicAdd(cnt, 0u) < gen) { __builtin_amdgcn_s_sleep(2); }
  }
  __syncthreads();
  __threadfence();                 // acquire: invalidate stale h in near caches
}

// ---------------------------------------------------------------------------
// Kernel 0: parallel prep — transpose x to [T][B][IN] bf16, convert W_lin.
// All fp32->bf16 VALU work happens here, OFF the serial recurrence path.
// ---------------------------------------------------------------------------
__global__ void prep_kernel(const float* __restrict__ x,     // [B,T,IN]
                            const float* __restrict__ Wlin,  // [64,256]
                            bf16_t* __restrict__ xt,         // [T,B,IN]
                            bf16_t* __restrict__ wl) {       // [64,256]
  const size_t stride = (size_t)gridDim.x * blockDim.x;
  size_t i0 = (size_t)blockIdx.x * blockDim.x + threadIdx.x;
  const size_t n = (size_t)B_ * T_ * IN_;
  for (size_t idx = i0; idx < n; idx += stride) {
    int k = (int)(idx & (IN_ - 1));
    size_t bt = idx >> 6;                 // = b*T + t
    int tt = (int)(bt & (T_ - 1));
    int bb = (int)(bt >> 10);
    xt[((size_t)tt * B_ + bb) * IN_ + k] = f2bf(x[idx]);
  }
  if (i0 < (size_t)OUT_ * H_) wl[i0] = f2bf(Wlin[i0]);
}

// ---------------------------------------------------------------------------
// Kernel 1: persistent 2-layer LSTM recurrence.
// grid = NWG blocks x 512 threads. Dynamic LDS = 312320 bytes (< 320 KB/WGP).
// ---------------------------------------------------------------------------
__global__ void lstm_persistent(const bf16_t* __restrict__ xt,    // [T,B,IN] bf16
                                const float* __restrict__ Wih0,   // [1024,64]
                                const float* __restrict__ Whh0,   // [1024,256]
                                const float* __restrict__ bih0,
                                const float* __restrict__ bhh0,
                                const float* __restrict__ Wih1,   // [1024,256]
                                const float* __restrict__ Whh1,   // [1024,256]
                                const float* __restrict__ bih1,
                                const float* __restrict__ bhh1,
                                bf16_t* __restrict__ h0buf,       // [2][128][256]
                                bf16_t* __restrict__ h1buf,       // [2][128][256]
                                bf16_t* __restrict__ h1all,       // [T][128][256]
                                unsigned* __restrict__ cnt) {
  extern __shared__ char smem[];
  bf16_t* sWih0  = (bf16_t*)(smem);                         // 128x64   16 KB
  bf16_t* sWhh0  = (bf16_t*)(smem + 16384);                 // 128x256  64 KB
  bf16_t* sWih1  = (bf16_t*)(smem + 16384 + 65536);         // 128x256  64 KB
  bf16_t* sWhh1  = (bf16_t*)(smem + 16384 + 2 * 65536);     // 128x256  64 KB
  float*  sGates = (float*)(smem + 16384 + 3 * 65536);      // 128x128  64 KB
  float*  sC0    = (float*)(smem + 16384 + 4 * 65536);      // 128x32   16 KB
  float*  sC1    = (float*)(smem + 16384 + 4 * 65536 + 16384);
  float*  sB0    = (float*)(smem + 16384 + 4 * 65536 + 2 * 16384);      // 128 f32
  float*  sB1    = (float*)(smem + 16384 + 4 * 65536 + 2 * 16384 + 512);

  const int tid  = threadIdx.x;
  const int wg   = blockIdx.x;           // owns hidden units [wg*32, wg*32+32)
  const int wave = tid >> 5;
  const int lane = tid & 31;
  const int wr   = wave >> 2;            // M-tile pair 0..3
  const int wc   = wave & 3;             // N-tile pair 0..3
  const int Mb   = wr * 32;
  const int Nb   = wc * 32;
  const int crow = (lane >> 4) << 3;     // C-tile: M = crow + r, N = lane&15
  const int ccol = lane & 15;

  unsigned gen = 0;

  // ---- stage weight slices (fp32 -> bf16) into LDS; local gate col n -> global
  //      gate row (n/32)*256 + wg*32 + n%32 (PyTorch i,f,g,o blocks) ----
  for (int idx = tid; idx < NGATE * IN_; idx += THR) {
    int n = idx >> 6, k = idx & 63;
    int g = ((n >> 5) << 8) + wg * 32 + (n & 31);
    sWih0[n * IN_ + k] = f2bf(Wih0[(size_t)g * IN_ + k]);
  }
  for (int idx = tid; idx < NGATE * H_; idx += THR) {
    int n = idx >> 8, k = idx & 255;
    int g = ((n >> 5) << 8) + wg * 32 + (n & 31);
    sWhh0[n * H_ + k] = f2bf(Whh0[(size_t)g * H_ + k]);
    sWih1[n * H_ + k] = f2bf(Wih1[(size_t)g * H_ + k]);
    sWhh1[n * H_ + k] = f2bf(Whh1[(size_t)g * H_ + k]);
  }
  // fused biases for this WG's gate slice: index q*32+j
  for (int idx = tid; idx < NGATE; idx += THR) {
    int g = ((idx >> 5) << 8) + wg * 32 + (idx & 31);
    sB0[idx] = bih0[g] + bhh0[g];
    sB1[idx] = bih1[g] + bhh1[g];
  }
  // zero cell state (this WG's slice) and h ping-pong buffers (all WGs write 0)
  for (int idx = tid; idx < B_ * 32; idx += THR) { sC0[idx] = 0.f; sC1[idx] = 0.f; }
  {
    bf16_t z = f2bf(0.f);
    for (int idx = tid; idx < 2 * B_ * H_; idx += THR) { h0buf[idx] = z; h1buf[idx] = z; }
  }
  grid_barrier(cnt, gen);

#pragma unroll 1
  for (int t = 0; t < T_; ++t) {
    const int cur = t & 1, nxt = cur ^ 1;
    const bf16_t* h0c = h0buf + (size_t)cur * B_ * H_;
    bf16_t*       h0n = h0buf + (size_t)nxt * B_ * H_;
    const bf16_t* h1c = h1buf + (size_t)cur * B_ * H_;
    bf16_t*       h1n = h1buf + (size_t)nxt * B_ * H_;
    const bf16_t* xb  = xt + (size_t)t * B_ * IN_;   // [128][64] bf16, contiguous

    // ================= layer 0: gates = x_t@Wih0^T + h0@Whh0^T ==============
    {
      v8f acc[2][2];
#pragma unroll
      for (int mi = 0; mi < 2; ++mi)
#pragma unroll
        for (int ni = 0; ni < 2; ++ni) acc[mi][ni] = vzero8();

#pragma unroll
      for (int kt = 0; kt < 2; ++kt) {                // K = 64 (input part)
        v16bf a0 = ld_bf_tile(xb, IN_, Mb,      kt * 32);
        v16bf a1 = ld_bf_tile(xb, IN_, Mb + 16, kt * 32);
#pragma unroll
        for (int ni = 0; ni < 2; ++ni) {
          v16bf b = ld_bf_tile(sWih0, IN_, Nb + ni * 16, kt * 32);
          acc[0][ni] = wmma_bf16(a0, b, acc[0][ni]);
          acc[1][ni] = wmma_bf16(a1, b, acc[1][ni]);
        }
      }
#pragma unroll
      for (int kt = 0; kt < 8; ++kt) {                // K = 256 (recurrent part)
        v16bf a0 = ld_bf_tile(h0c, H_, Mb,      kt * 32);
        v16bf a1 = ld_bf_tile(h0c, H_, Mb + 16, kt * 32);
#pragma unroll
        for (int ni = 0; ni < 2; ++ni) {
          v16bf b = ld_bf_tile(sWhh0, H_, Nb + ni * 16, kt * 32);
          acc[0][ni] = wmma_bf16(a0, b, acc[0][ni]);
          acc[1][ni] = wmma_bf16(a1, b, acc[1][ni]);
        }
      }
#pragma unroll
      for (int mi = 0; mi < 2; ++mi)
#pragma unroll
        for (int ni = 0; ni < 2; ++ni)
#pragma unroll
          for (int r = 0; r < 8; ++r)
            sGates[(Mb + mi * 16 + crow + r) * NGATE + (Nb + ni * 16 + ccol)] = acc[mi][ni][r];
    }
    __syncthreads();

    // elementwise LSTM cell for this WG's 32 hidden units
#pragma unroll 1
    for (int it = 0; it < (B_ * 32) / THR; ++it) {
      int e = (it << 9) + tid;
      int m = e >> 5, j = e & 31;
      float xi = sGates[m * NGATE + j]      + sB0[j];
      float xf = sGates[m * NGATE + 32 + j] + sB0[32 + j];
      float xg = sGates[m * NGATE + 64 + j] + sB0[64 + j];
      float xo = sGates[m * NGATE + 96 + j] + sB0[96 + j];
      float c = sigmoidf_(xf) * sC0[m * 32 + j] + sigmoidf_(xi) * tanhf_(xg);
      float h = sigmoidf_(xo) * tanhf_(c);
      sC0[m * 32 + j] = c;
      h0n[(size_t)m * H_ + wg * 32 + j] = f2bf(h);
    }
    grid_barrier(cnt, gen);    // all WGs now see full h0_new

    // prefetch next step's x tile (16 KB) while layer 1 runs
    if (t + 1 < T_ && tid < 128) {
      const bf16_t* nx = xt + (size_t)(t + 1) * B_ * IN_;
      __builtin_prefetch(nx + (size_t)tid * 64, 0, 0);   // 128B granules
    }

    // ================= layer 1: gates = h0_new@Wih1^T + h1@Whh1^T ===========
    {
      v8f acc[2][2];
#pragma unroll
      for (int mi = 0; mi < 2; ++mi)
#pragma unroll
        for (int ni = 0; ni < 2; ++ni) acc[mi][ni] = vzero8();

#pragma unroll
      for (int kt = 0; kt < 8; ++kt) {                // h0_new part
        v16bf a0 = ld_bf_tile(h0n, H_, Mb,      kt * 32);
        v16bf a1 = ld_bf_tile(h0n, H_, Mb + 16, kt * 32);
#pragma unroll
        for (int ni = 0; ni < 2; ++ni) {
          v16bf b = ld_bf_tile(sWih1, H_, Nb + ni * 16, kt * 32);
          acc[0][ni] = wmma_bf16(a0, b, acc[0][ni]);
          acc[1][ni] = wmma_bf16(a1, b, acc[1][ni]);
        }
      }
#pragma unroll
      for (int kt = 0; kt < 8; ++kt) {                // h1 recurrent part
        v16bf a0 = ld_bf_tile(h1c, H_, Mb,      kt * 32);
        v16bf a1 = ld_bf_tile(h1c, H_, Mb + 16, kt * 32);
#pragma unroll
        for (int ni = 0; ni < 2; ++ni) {
          v16bf b = ld_bf_tile(sWhh1, H_, Nb + ni * 16, kt * 32);
          acc[0][ni] = wmma_bf16(a0, b, acc[0][ni]);
          acc[1][ni] = wmma_bf16(a1, b, acc[1][ni]);
        }
      }
#pragma unroll
      for (int mi = 0; mi < 2; ++mi)
#pragma unroll
        for (int ni = 0; ni < 2; ++ni)
#pragma unroll
          for (int r = 0; r < 8; ++r)
            sGates[(Mb + mi * 16 + crow + r) * NGATE + (Nb + ni * 16 + ccol)] = acc[mi][ni][r];
    }
    __syncthreads();

#pragma unroll 1
    for (int it = 0; it < (B_ * 32) / THR; ++it) {
      int e = (it << 9) + tid;
      int m = e >> 5, j = e & 31;
      float xi = sGates[m * NGATE + j]      + sB1[j];
      float xf = sGates[m * NGATE + 32 + j] + sB1[32 + j];
      float xg = sGates[m * NGATE + 64 + j] + sB1[64 + j];
      float xo = sGates[m * NGATE + 96 + j] + sB1[96 + j];
      float c = sigmoidf_(xf) * sC1[m * 32 + j] + sigmoidf_(xi) * tanhf_(xg);
      float h = sigmoidf_(xo) * tanhf_(c);
      sC1[m * 32 + j] = c;
      bf16_t hb = f2bf(h);
      h1n[(size_t)m * H_ + wg * 32 + j] = hb;
      h1all[((size_t)t * B_ + m) * H_ + wg * 32 + j] = hb;   // spill for projection
    }
    grid_barrier(cnt, gen);    // h1_new visible before next step
  }
}

// ---------------------------------------------------------------------------
// Kernel 2: out = h1_all @ W_lin^T + b_lin  (M = T*B rows t-major, N=64, K=256)
// grid = 1024 blocks x 256 threads (8 waves, each one 16x64 output tile)
// ---------------------------------------------------------------------------
__global__ void proj_kernel(const bf16_t* __restrict__ h1all,   // [T*B, 256]
                            const bf16_t* __restrict__ wl,      // [64, 256] bf16
                            const float* __restrict__ blin,     // [64]
                            float* __restrict__ out) {          // [B, T, 64]
  __shared__ bf16_t sW[OUT_ * H_];                               // 32 KB
  const int tid = threadIdx.x;
  for (int idx = tid; idx < OUT_ * H_; idx += 256) sW[idx] = wl[idx];
  __syncthreads();

  const int wave = tid >> 5;
  const int lane = tid & 31;
  const int row0 = (blockIdx.x * 8 + wave) * 16;

  v8f acc[4];
#pragma unroll
  for (int nt = 0; nt < 4; ++nt) acc[nt] = vzero8();

#pragma unroll
  for (int kt = 0; kt < 8; ++kt) {
    v16bf a = ld_bf_tile(h1all, H_, row0, kt * 32);
#pragma unroll
    for (int nt = 0; nt < 4; ++nt) {
      v16bf b = ld_bf_tile(sW, H_, nt * 16, kt * 32);
      acc[nt] = wmma_bf16(a, b, acc[nt]);
    }
  }

  const int crow = (lane >> 4) << 3;
  const int ccol = lane & 15;
#pragma unroll
  for (int nt = 0; nt < 4; ++nt) {
    int n = nt * 16 + ccol;
    float bias = blin[n];
#pragma unroll
    for (int r = 0; r < 8; ++r) {
      int row = row0 + crow + r;             // row = t*128 + b
      int tt = row >> 7;
      int bb = row & 127;
      out[((size_t)bb * T_ + tt) * OUT_ + n] = acc[nt][r] + bias;
    }
  }
}

// ---------------------------------------------------------------------------
extern "C" void kernel_launch(void* const* d_in, const int* in_sizes, int n_in,
                              void* d_out, int out_size, void* d_ws, size_t ws_size,
                              hipStream_t stream) {
  const float* x    = (const float*)d_in[0];
  const float* Wih0 = (const float*)d_in[1];
  const float* Whh0 = (const float*)d_in[2];
  const float* bih0 = (const float*)d_in[3];
  const float* bhh0 = (const float*)d_in[4];
  const float* Wih1 = (const float*)d_in[5];
  const float* Whh1 = (const float*)d_in[6];
  const float* bih1 = (const float*)d_in[7];
  const float* bhh1 = (const float*)d_in[8];
  const float* Wlin = (const float*)d_in[9];
  const float* blin = (const float*)d_in[10];

  char* ws = (char*)d_ws;
  unsigned* cnt  = (unsigned*)ws;                                   // barrier counter
  bf16_t* h0buf  = (bf16_t*)(ws + 1024);                            // 128 KB
  bf16_t* h1buf  = (bf16_t*)(ws + 1024 + 131072);                   // 128 KB
  bf16_t* xt     = (bf16_t*)(ws + 1024 + 262144);                   // 16 MB
  bf16_t* wl     = (bf16_t*)(ws + 1024 + 262144 + 16777216);        // 32 KB
  bf16_t* h1all  = (bf16_t*)(ws + 1024 + 262144 + 16777216 + 32768);// 64 MB

  const size_t ldsBytes = 16384 + 4 * 65536 + 2 * 16384 + 1024;     // 312320 B
  hipFuncSetAttribute((const void*)lstm_persistent,
                      hipFuncAttributeMaxDynamicSharedMemorySize, (int)ldsBytes);

  hipMemsetAsync(cnt, 0, 64, stream);                               // reset grid barrier

  prep_kernel<<<4096, 256, 0, stream>>>(x, Wlin, xt, wl);

  lstm_persistent<<<NWG, THR, ldsBytes, stream>>>(
      xt, Wih0, Whh0, bih0, bhh0, Wih1, Whh1, bih1, bhh1,
      h0buf, h1buf, h1all, cnt);

  proj_kernel<<<(T_ * B_) / 128, 256, 0, stream>>>(h1all, wl, blin, (float*)d_out);
}
